// Nat2D_11519102288428
// MI455X (gfx1250) — matrix operand order repaired and checked by reading, not compiled, not run on hardware
//
#include <hip/hip_runtime.h>

// NAT2D: qkv GEMM (bf16 WMMA) -> 7x7 neighborhood attention (VALU, online softmax)
//        -> proj GEMM (bf16 WMMA). All intermediates in d_ws.

typedef __bf16 bf16x16 __attribute__((ext_vector_type(16)));
typedef float  floatx8 __attribute__((ext_vector_type(8)));

#define HEADS     8
#define HEAD_DIM  64
#define DIM       512
#define HH        80
#define WW        80
#define KWIN      7
#define NPIX      (HH * WW)

__device__ __forceinline__ unsigned short f2bf(float f) {
  union { float f; unsigned u; } x; x.f = f;
  unsigned u = x.u;
  u += 0x7FFFu + ((u >> 16) & 1u);   // round-to-nearest-even
  return (unsigned short)(u >> 16);
}

__global__ void cvt_f32_bf16(const float* __restrict__ in,
                             unsigned short* __restrict__ out, int n) {
  int i = blockIdx.x * blockDim.x + threadIdx.x;
  if (i < n) out[i] = f2bf(in[i]);
}

union FragAB { bf16x16 v; float4 f4[2]; };

// C[M][N] = A[M][K] @ B[N][K]^T + bias[N]; A,B bf16 (K contiguous), C f32.
// grid = (N/128, M/128), block = 256 (8 waves; 4 in M x 2 in N).
// Each wave: 32x64 tile = 2 A-frags x 4 B-frags of v_wmma_f32_16x16x32_bf16.
__global__ __launch_bounds__(256)
void gemm_bf16_wmma(const unsigned short* __restrict__ A,
                    const unsigned short* __restrict__ B,
                    const float* __restrict__ bias,
                    float* __restrict__ C,
                    int M, int N, int K) {
  const int lane = threadIdx.x & 31;
  const int wave = threadIdx.x >> 5;
  const int wm   = wave >> 1;                   // 0..3
  const int wn   = wave & 1;                    // 0..1
  const int m0   = blockIdx.y * 128 + wm * 32;
  const int n0   = blockIdx.x * 128 + wn * 64;
  const int lh   = lane & 15;
  const int ka   = (lane >> 4) * 8;             // A-frag K offset per half-wave
  const int kb   = (lane >> 4) * 8;             // B-frag K offset per half-wave

  floatx8 acc[2][4];
#pragma unroll
  for (int i = 0; i < 2; ++i)
#pragma unroll
    for (int j = 0; j < 4; ++j) {
      floatx8 z = {0.f, 0.f, 0.f, 0.f, 0.f, 0.f, 0.f, 0.f};
      acc[i][j] = z;
    }

  for (int k = 0; k < K; k += 32) {
    FragAB a[2], b[4];
#pragma unroll
    for (int i = 0; i < 2; ++i) {
      const unsigned short* pa = A + (size_t)(m0 + 16 * i + lh) * K + k + ka;
      a[i].f4[0] = *(const float4*)(pa);        // K = ka .. ka+7
      a[i].f4[1] = *(const float4*)(pa + 16);   // K = ka+16 .. ka+23
    }
#pragma unroll
    for (int j = 0; j < 4; ++j) {
      const unsigned short* pb = B + (size_t)(n0 + 16 * j + lh) * K + k + kb;
      b[j].f4[0] = *(const float4*)(pb);        // K = kb .. kb+7
      b[j].f4[1] = *(const float4*)(pb + 16);   // K = kb+16 .. kb+23
    }
#pragma unroll
    for (int i = 0; i < 2; ++i)
#pragma unroll
      for (int j = 0; j < 4; ++j)
        acc[i][j] = __builtin_amdgcn_wmma_f32_16x16x32_bf16(
            false, a[i].v, false, b[j].v, (short)0, acc[i][j], false, false);
  }

  const int rbase = (lane >> 4) * 8;
#pragma unroll
  for (int i = 0; i < 2; ++i)
#pragma unroll
    for (int j = 0; j < 4; ++j) {
      const int col = n0 + 16 * j + lh;
      const float bv = bias[col];
#pragma unroll
      for (int r = 0; r < 8; ++r) {
        const int row = m0 + 16 * i + rbase + r;
        C[(size_t)row * N + col] = acc[i][j][r] + bv;
      }
    }
}

// One thread per (pixel, head). qkv layout: [NPIX][3*DIM] with cols
// [0,512)=q, [512,1024)=k, [1024,1536)=v; within each, head*64+d.
// Online-softmax over the 49 neighbors; emits bf16 [NPIX][DIM].
__global__ __launch_bounds__(256)
void nat2d_attn(const float* __restrict__ qkv,
                const float* __restrict__ rpb,           // [HEADS][13][13]
                unsigned short* __restrict__ outb) {
  const int t    = blockIdx.x * blockDim.x + threadIdx.x;
  const int head = t & (HEADS - 1);
  const int p    = t >> 3;
  const int i    = p / WW;
  const int j    = p % WW;

  int si = i - (KWIN / 2); if (si < 0) si = 0; if (si > HH - KWIN) si = HH - KWIN;
  int sj = j - (KWIN / 2); if (sj < 0) sj = 0; if (sj > WW - KWIN) sj = WW - KWIN;

  const float* qrow = qkv + (size_t)p * (3 * DIM) + head * HEAD_DIM;
  float q[HEAD_DIM];
#pragma unroll
  for (int d = 0; d < HEAD_DIM; d += 4) {
    float4 v = *(const float4*)(qrow + d);
    q[d]   = v.x * 0.125f;  q[d+1] = v.y * 0.125f;   // hd^-0.5 = 1/8
    q[d+2] = v.z * 0.125f;  q[d+3] = v.w * 0.125f;
  }

  const float* rp = rpb + head * (13 * 13);
  float m = -3.402823466e38f;
  float l = 0.f;
  float acc[HEAD_DIM];
#pragma unroll
  for (int d = 0; d < HEAD_DIM; ++d) acc[d] = 0.f;

  for (int a = 0; a < KWIN; ++a) {
    for (int c = 0; c < KWIN; ++c) {
      const int np = (si + a) * WW + (sj + c);
      const float* krow = qkv + (size_t)np * (3 * DIM) + DIM + head * HEAD_DIM;
      float s = 0.f;
#pragma unroll
      for (int d = 0; d < HEAD_DIM; d += 4) {
        float4 kv = *(const float4*)(krow + d);
        s += q[d] * kv.x + q[d+1] * kv.y + q[d+2] * kv.z + q[d+3] * kv.w;
      }
      s += rp[(si + a - i + 6) * 13 + (sj + c - j + 6)];

      const float mnew  = fmaxf(m, s);
      const float scale = __expf(m - mnew);     // 0 on first iteration
      const float e     = __expf(s - mnew);
      l = l * scale + e;
      const float* vrow = qkv + (size_t)np * (3 * DIM) + 2 * DIM + head * HEAD_DIM;
#pragma unroll
      for (int d = 0; d < HEAD_DIM; d += 4) {
        float4 vv = *(const float4*)(vrow + d);
        acc[d]   = acc[d]   * scale + e * vv.x;
        acc[d+1] = acc[d+1] * scale + e * vv.y;
        acc[d+2] = acc[d+2] * scale + e * vv.z;
        acc[d+3] = acc[d+3] * scale + e * vv.w;
      }
      m = mnew;
    }
  }

  const float inv = 1.0f / l;
  unsigned short* orow = outb + (size_t)p * DIM + head * HEAD_DIM;
#pragma unroll
  for (int d = 0; d < HEAD_DIM; ++d) orow[d] = f2bf(acc[d] * inv);
}

extern "C" void kernel_launch(void* const* d_in, const int* in_sizes, int n_in,
                              void* d_out, int out_size, void* d_ws, size_t ws_size,
                              hipStream_t stream) {
  const float* x      = (const float*)d_in[0];   // [6400][512]
  const float* qkv_w  = (const float*)d_in[1];   // [1536][512]
  const float* qkv_b  = (const float*)d_in[2];   // [1536]
  const float* rpb    = (const float*)d_in[3];   // [8][13][13]
  const float* proj_w = (const float*)d_in[4];   // [512][512]
  const float* proj_b = (const float*)d_in[5];   // [512]
  float* out = (float*)d_out;                    // [6400][512]

  char* ws = (char*)d_ws;
  // Workspace layout (all offsets 16B aligned):
  unsigned short* xb     = (unsigned short*)(ws + 0);              //  6,553,600 B
  unsigned short* qkvwb  = (unsigned short*)(ws + 6553600);        //  1,572,864 B
  unsigned short* projwb = (unsigned short*)(ws + 8126464);        //    524,288 B
  float*          qkvf   = (float*)         (ws + 8650752);        // 39,321,600 B
  unsigned short* aoutb  = (unsigned short*)(ws + 47972352);       //  6,553,600 B
  // total: 54,525,952 B

  const int n_x  = NPIX * DIM;        // 3,276,800
  const int n_qw = 3 * DIM * DIM;     //   786,432
  const int n_pw = DIM * DIM;         //   262,144

  cvt_f32_bf16<<<(n_x  + 255) / 256, 256, 0, stream>>>(x,      xb,     n_x);
  cvt_f32_bf16<<<(n_qw + 255) / 256, 256, 0, stream>>>(qkv_w,  qkvwb,  n_qw);
  cvt_f32_bf16<<<(n_pw + 255) / 256, 256, 0, stream>>>(proj_w, projwb, n_pw);

  // QKV GEMM: M=6400, N=1536, K=512
  gemm_bf16_wmma<<<dim3(1536 / 128, 6400 / 128), 256, 0, stream>>>(
      xb, qkvwb, qkv_b, qkvf, NPIX, 3 * DIM, DIM);

  // Neighborhood attention: 6400 pixels x 8 heads = 51200 threads
  nat2d_attn<<<dim3((NPIX * HEADS) / 256), 256, 0, stream>>>(qkvf, rpb, aoutb);

  // Proj GEMM: M=6400, N=512, K=512
  gemm_bf16_wmma<<<dim3(512 / 128, 6400 / 128), 256, 0, stream>>>(
      aoutb, projwb, proj_b, out, NPIX, DIM, DIM);
}
